// lstm_seq2seq_36206574305735
// MI455X (gfx1250) — matrix operand order repaired and checked by reading, not compiled, not run on hardware
//
#include <hip/hip_runtime.h>
#include <hip/hip_bf16.h>

// ---------------------------------------------------------------------------
// Seq2seq stacked LSTM for MI455X (gfx1250, wave32, WMMA).
//
// Strategy: the recurrence is independent per batch row, so each workgroup
// owns a batch tile (BH rows) and runs ALL 120 encoder + 120 decoder steps
// internally (no inter-workgroup traffic). Gate GEMMs use
// v_wmma_f32_16x16x32_f16 with K=128 ([x_pad(64) | h(64)]), weights
// pre-swizzled once per launch into per-lane-contiguous fragment order.
// ---------------------------------------------------------------------------

typedef _Float16 v8h  __attribute__((ext_vector_type(8)));
typedef _Float16 v16h __attribute__((ext_vector_type(16)));
typedef float    v8f  __attribute__((ext_vector_type(8)));

#define SEQ_LEN   120
#define BATCH     2048
#define NIN       6
#define HID       64
#define LAYERS    4
#define TGT_LEN   120

#define BH        32                 // batch rows per workgroup
#define NWAVES    16                 // one wave per 16-column gate tile (N=256)
#define NTHREADS  (NWAVES * 32)
#define AP        88                 // f16 activation row pitch (176B, 16B-aligned)
#define GP        260                // gates row pitch in floats (breaks bank alignment)

// fragment-ordered weights: [phase][layer][kc(4)][ntile(16)][lane(32)][16] halves
#define FRAG_HALVES_PER_PHASE (LAYERS * 4 * 16 * 32 * 16)   // 131072 halves = 256KB

// ---------------------------------------------------------------------------
// Prep: swizzle fp32 weights -> f16 fragments so the main kernel's B-matrix
// fragment is a single contiguous 32B load per lane.
//   K layout per layer: rows 0..63 = Wih^T (layer0: cols 0..5 valid, rest 0),
//                       rows 64..127 = Whh^T.
//   B fragment (32x16, 16-bit): lane n = lane&15 (column), K =
//                       kc*32 + (lane>>4)*8 + {0..7, 16..23}.
// ---------------------------------------------------------------------------
__global__ __launch_bounds__(256) void prep_weights(
    const float* __restrict__ eWih0, const float* __restrict__ eWih,
    const float* __restrict__ eWhh,
    const float* __restrict__ dWih0, const float* __restrict__ dWih,
    const float* __restrict__ dWhh,
    _Float16* __restrict__ frag)
{
  int idx   = blockIdx.x * 256 + threadIdx.x;     // 0 .. 2*131072-1
  int i     =  idx        & 15;
  int lane  = (idx >> 4)  & 31;
  int nt    = (idx >> 9)  & 15;
  int kc    = (idx >> 13) & 3;
  int l     = (idx >> 15) & 3;
  int phase = (idx >> 17) & 1;                    // 0 = encoder, 1 = decoder

  const float* Wih0 = phase ? dWih0 : eWih0;
  const float* Wih  = phase ? dWih  : eWih;
  const float* Whh  = phase ? dWhh  : eWhh;

  int n = nt * 16 + (lane & 15);                  // gate column (4H index)
  int k = kc * 32 + ((lane >> 4) << 3) + (i & 7) + ((i >> 3) << 4); // combined K index

  float v;
  if (k < HID) {                                  // x-part
    v = (l == 0) ? ((k < NIN) ? Wih0[n * NIN + k] : 0.f)
                 : Wih[(((l - 1) * 4 * HID + n) * HID) + k];
  } else {                                        // h-part
    v = Whh[((l * 4 * HID + n) * HID) + (k - HID)];
  }
  frag[idx] = (_Float16)v;
}

// ---------------------------------------------------------------------------
// A-matrix fragment load (16x32 f16, documented VGPR layout):
//   lane<16:  K = {kb.. kb+7, kb+16.. kb+23} with kb = kc*32
//   lane>=16: same shifted by +8.
// Two contiguous 16B LDS loads per lane.
// ---------------------------------------------------------------------------
__device__ __forceinline__ v16h a_frag(const _Float16* buf, int ms, int kl, int lane)
{
  int m  = ms * 16 + (lane & 15);
  int kb = kl * 32 + ((lane >> 4) << 3);
  const _Float16* p = buf + m * AP + kb;
  v8h lo = *(const v8h*)(p);
  v8h hi = *(const v8h*)(p + 16);
  return __builtin_shufflevector(lo, hi, 0,1,2,3,4,5,6,7,8,9,10,11,12,13,14,15);
}

__device__ __forceinline__ float sigm(float x) { return 1.f / (1.f + __expf(-x)); }

// ---------------------------------------------------------------------------
// Persistent per-batch-tile kernel: full encoder + decoder recurrence.
// ---------------------------------------------------------------------------
__global__ __launch_bounds__(NTHREADS) void seq2seq_kernel(
    const float*    __restrict__ x,
    const _Float16* __restrict__ encFrag,
    const _Float16* __restrict__ decFrag,
    const float*    __restrict__ enc_b,
    const float*    __restrict__ dec_b,
    const float*    __restrict__ linW,
    const float*    __restrict__ linb,
    float*          __restrict__ out)
{
  extern __shared__ char smem[];
  _Float16* inp  = (_Float16*)smem;                                  // [BH][AP]
  _Float16* hbuf = inp + BH * AP;                                    // [L][BH][AP]
  float* gates = (float*)(smem + (size_t)(1 + LAYERS) * BH * AP * 2);// [BH][GP]
  float* cbuf  = gates + BH * GP;                                    // [L][BH][HID]
  float* hf32  = cbuf + LAYERS * BH * HID;                           // [BH][HID]
  float* outb  = hf32 + BH * HID;                                    // [BH][NIN]

  const int tid  = threadIdx.x;
  const int lane = tid & 31;
  const int wave = tid >> 5;                  // n-tile (0..15)
  const int m0   = blockIdx.x * BH;
  const int ncol = wave * 16 + (lane & 15);   // this lane's gate column

  float biasE[LAYERS], biasD[LAYERS];
#pragma unroll
  for (int l = 0; l < LAYERS; ++l) {
    biasE[l] = enc_b[l * 4 * HID + ncol];
    biasD[l] = dec_b[l * 4 * HID + ncol];
  }

  // zero-init c and h state
  for (int e = tid; e < LAYERS * BH * HID; e += NTHREADS) cbuf[e] = 0.f;
  for (int e = tid; e < BH * HID; e += NTHREADS) {
    int m = e >> 6, j = e & 63;
#pragma unroll
    for (int l = 0; l < LAYERS; ++l) hbuf[l * BH * AP + m * AP + j] = (_Float16)0.f;
  }

  // one stacked-LSTM timestep (inp already holds padded f16 input)
  auto step = [&](const _Float16* wfrag, const float* bias) {
#pragma unroll 1
    for (int l = 0; l < LAYERS; ++l) {
      _Float16* hl = hbuf + l * BH * AP;
      // hoist this wave's 4 B-fragments (contiguous 32B per lane, L2-hot)
      const _Float16* wf = wfrag + ((size_t)((l * 4) * 16 + wave) * 32 + lane) * 16;
      const size_t kcs = (size_t)16 * 32 * 16;   // kc stride in halves
      v16h b0 = *(const v16h*)(wf + 0 * kcs);
      v16h b1 = *(const v16h*)(wf + 1 * kcs);
      v16h b2 = *(const v16h*)(wf + 2 * kcs);
      v16h b3 = *(const v16h*)(wf + 3 * kcs);
      float bs = bias[l];
#pragma unroll
      for (int ms = 0; ms < BH / 16; ++ms) {
        v8f acc = { bs, bs, bs, bs, bs, bs, bs, bs };
        acc = __builtin_amdgcn_wmma_f32_16x16x32_f16(false, a_frag(inp, ms, 0, lane),
                                                     false, b0, (short)0, acc, false, false);
        acc = __builtin_amdgcn_wmma_f32_16x16x32_f16(false, a_frag(inp, ms, 1, lane),
                                                     false, b1, (short)0, acc, false, false);
        acc = __builtin_amdgcn_wmma_f32_16x16x32_f16(false, a_frag(hl, ms, 0, lane),
                                                     false, b2, (short)0, acc, false, false);
        acc = __builtin_amdgcn_wmma_f32_16x16x32_f16(false, a_frag(hl, ms, 1, lane),
                                                     false, b3, (short)0, acc, false, false);
        int row0 = ms * 16 + ((lane >> 4) << 3);
#pragma unroll
        for (int v = 0; v < 8; ++v) gates[(row0 + v) * GP + ncol] = acc[v];
      }
      __syncthreads();

      // elementwise cell update (fp32 state), h -> f16 for next matmuls
      float* cl = cbuf + l * BH * HID;
      for (int e = tid; e < BH * HID; e += NTHREADS) {
        int m = e >> 6, j = e & 63;
        const float* g = gates + m * GP + j;
        float ig = g[0], fg = g[HID], gg = g[2 * HID], og = g[3 * HID];
        float c = sigm(fg) * cl[e] + sigm(ig) * tanhf(gg);
        float h = sigm(og) * tanhf(c);
        cl[e] = c;
        _Float16 hh = (_Float16)h;
        hl[m * AP + j]  = hh;   // recurrent state for this layer
        inp[m * AP + j] = hh;   // input to next layer
        if (l == LAYERS - 1) hf32[e] = h;
      }
      __syncthreads();
    }
  };

  // ---- encoder ----
  for (int t = 0; t < SEQ_LEN; ++t) {
    const float* xt = x + ((size_t)t * BATCH + m0) * NIN;
    for (int e = tid; e < BH * HID; e += NTHREADS) {
      int m = e >> 6, j = e & 63;
      inp[m * AP + j] = (j < NIN) ? (_Float16)xt[m * NIN + j] : (_Float16)0.f;
    }
    __syncthreads();
    step(encFrag, biasE);
  }

  // ---- decoder: seed with last observed frame ----
  {
    const float* xt = x + ((size_t)(SEQ_LEN - 1) * BATCH + m0) * NIN;
    for (int e = tid; e < BH * HID; e += NTHREADS) {
      int m = e >> 6, j = e & 63;
      inp[m * AP + j] = (j < NIN) ? (_Float16)xt[m * NIN + j] : (_Float16)0.f;
    }
    __syncthreads();
  }
  for (int t = 0; t < TGT_LEN; ++t) {
    step(decFrag, biasD);
    // linear head: out[m][j] = hf32[m][:] . linW[j][:] + linb[j]
    for (int e = tid; e < BH * NIN; e += NTHREADS) {
      int m = e / NIN, j = e - m * NIN;
      float s = linb[j];
      const float* hv = hf32 + m * HID;
      const float* wj = linW + j * HID;
#pragma unroll 8
      for (int k = 0; k < HID; ++k) s += hv[k] * wj[k];
      out[((size_t)t * BATCH + m0 + m) * NIN + j] = s;
      outb[e] = s;
    }
    __syncthreads();
    // feedback: next step input = out, zero-padded to 64
    for (int e = tid; e < BH * HID; e += NTHREADS) {
      int m = e >> 6, j = e & 63;
      inp[m * AP + j] = (j < NIN) ? (_Float16)outb[m * NIN + j] : (_Float16)0.f;
    }
    __syncthreads();
  }
}

// ---------------------------------------------------------------------------
// Host side
// ---------------------------------------------------------------------------
extern "C" void kernel_launch(void* const* d_in, const int* in_sizes, int n_in,
                              void* d_out, int out_size, void* d_ws, size_t ws_size,
                              hipStream_t stream)
{
  (void)in_sizes; (void)n_in; (void)out_size; (void)ws_size;
  const float* x     = (const float*)d_in[0];
  const float* eWih0 = (const float*)d_in[1];
  const float* eWih  = (const float*)d_in[2];
  const float* eWhh  = (const float*)d_in[3];
  const float* eb    = (const float*)d_in[4];
  const float* dWih0 = (const float*)d_in[5];
  const float* dWih  = (const float*)d_in[6];
  const float* dWhh  = (const float*)d_in[7];
  const float* db    = (const float*)d_in[8];
  const float* linW  = (const float*)d_in[9];
  const float* linb  = (const float*)d_in[10];

  _Float16* frag = (_Float16*)d_ws;   // 2 * 256KB of fragment-ordered f16 weights

  prep_weights<<<(2 * FRAG_HALVES_PER_PHASE) / 256, 256, 0, stream>>>(
      eWih0, eWih, eWhh, dWih0, dWih, dWhh, frag);

  size_t lds = (size_t)(1 + LAYERS) * BH * AP * 2      // inp + hbuf (f16)
             + (size_t)BH * GP * 4                     // gates (f32)
             + (size_t)LAYERS * BH * HID * 4           // cbuf (f32)
             + (size_t)BH * HID * 4                    // hf32 (f32)
             + (size_t)BH * NIN * 4;                   // outb (f32)

  seq2seq_kernel<<<BATCH / BH, NTHREADS, lds, stream>>>(
      x, frag, frag + FRAG_HALVES_PER_PHASE, eb, db, linW, linb, (float*)d_out);
}
